// RLSTM_43310450213597
// MI455X (gfx1250) — matrix operand order repaired and checked by reading, compiled, not run on hardware
//
#include <hip/hip_runtime.h>

typedef __bf16 bf16_t;
typedef __bf16 v16bf __attribute__((ext_vector_type(16)));
typedef float  v8f   __attribute__((ext_vector_type(8)));

#if defined(__has_builtin)
#if __has_builtin(__builtin_amdgcn_sched_barrier)
#define SCHED_FENCE() __builtin_amdgcn_sched_barrier(0)
#endif
#endif
#ifndef SCHED_FENCE
#define SCHED_FENCE()
#endif

#define BATCH 256
#define SEQ   128
#define ISZ   512
#define HSZ   1024
#define OSZ   32
#define KTOT  (ISZ + HSZ)     // 1536
#define NTOT  (4 * HSZ)       // 4096
#define KT    (KTOT / 32)     // 48 k-tiles
#define NTL   (NTOT / 16)     // 256 n-tiles
#define MT    (BATCH / 16)    // 16 m-tiles
#define FRAG  512             // elements per 16x32 / 32x16 fragment

// ---------------------------------------------------------------------------
// Pack 8 f32 weight matrices into bf16 WMMA B-fragment layout.
// B[k][n]: k<512 -> w_i*[n_h*512+k], k>=512 -> w_h*[n_h*1024+(k-512)]
// Gate order along n: 0=i, 1=f, 2=c(candidate), 3=o.
// B-frag layout (16-bit, 32x16): N=lane%16, half=lane/16;
//                                K_local=(e/8)*16 + half*8 + e%8.
// ---------------------------------------------------------------------------
__global__ void k_pack_weights(const float* __restrict__ w_ii, const float* __restrict__ w_hi,
                               const float* __restrict__ w_if, const float* __restrict__ w_hf,
                               const float* __restrict__ w_ic, const float* __restrict__ w_hc,
                               const float* __restrict__ w_io, const float* __restrict__ w_ho,
                               bf16_t* __restrict__ Bpack) {
    int idx    = blockIdx.x * 256 + threadIdx.x;   // exactly KT*NTL*FRAG threads
    int f      = idx >> 9;
    int within = idx & 511;
    int lane   = within >> 4;
    int e      = within & 15;
    int kt = f / NTL;
    int nt = f % NTL;
    int K  = kt * 32 + ((e >> 3) << 4) + ((lane >> 4) << 3) + (e & 7);
    int n  = nt * 16 + (lane & 15);
    int g  = n >> 10;
    int nh = n & 1023;
    float v;
    if (K < ISZ) {
        const float* w = (g == 0) ? w_ii : ((g == 1) ? w_if : ((g == 2) ? w_ic : w_io));
        v = w[nh * ISZ + K];
    } else {
        const float* w = (g == 0) ? w_hi : ((g == 1) ? w_hf : ((g == 2) ? w_hc : w_ho));
        v = w[nh * HSZ + (K - ISZ)];
    }
    Bpack[idx] = (bf16_t)v;
}

__global__ void k_pack_bias(const float* __restrict__ b_ii, const float* __restrict__ b_hi,
                            const float* __restrict__ b_if, const float* __restrict__ b_hf,
                            const float* __restrict__ b_ic, const float* __restrict__ b_hc,
                            const float* __restrict__ b_io, const float* __restrict__ b_ho,
                            float* __restrict__ bias) {
    int n = blockIdx.x * 256 + threadIdx.x;   // 4096 threads
    int g  = n >> 10;
    int nh = n & 1023;
    float v;
    if      (g == 0) v = b_ii[nh] + b_hi[nh];
    else if (g == 1) v = b_if[nh] + b_hf[nh];
    else if (g == 2) v = b_ic[nh] + b_hc[nh];
    else             v = b_io[nh] + b_ho[nh];
    bias[n] = v;
}

__global__ void k_init(bf16_t* __restrict__ Apack, float* __restrict__ c) {
    int idx = blockIdx.x * 256 + threadIdx.x;      // MT*KT*FRAG = 393216 threads
    Apack[idx] = (bf16_t)0.0f;                     // zeros h-region (x overwritten later)
    if (idx < BATCH * HSZ) c[idx] = 0.0f;
}

// ---------------------------------------------------------------------------
// A-frag (16-bit, 16x32): lane = half*16 + M_local;
//   K_local = (v/4)*16 + half*8 + (v%4)*2 + p, e = 2v+p.
// ---------------------------------------------------------------------------
__device__ __forceinline__ int a_frag_index(int b, int K) {
    int kt = K >> 5, kl = K & 31;
    int mt = b >> 4, ml = b & 15;
    int half = (kl >> 3) & 1;
    int v    = ((kl >> 4) << 2) | ((kl >> 1) & 3);
    int p    = kl & 1;
    int e    = (v << 1) | p;
    int lane = (half << 4) | ml;
    return ((mt * KT + kt) << 9) + (lane << 4) + e;
}

// Pack x_t (f32) into A-fragment bf16 layout for k-tiles 0..15 (primes t=0).
__global__ void k_pack_x(const float* __restrict__ inputs, bf16_t* __restrict__ Apack, int t) {
    int idx = blockIdx.x * 256 + threadIdx.x;  // BATCH*ISZ = 131072 threads
    int b = idx >> 9;
    int k = idx & 511;
    Apack[a_frag_index(b, k)] = (bf16_t)inputs[(b * SEQ + t) * ISZ + k];
}

// ---------------------------------------------------------------------------
// Gate GEMM: [256 x 1536] x [1536 x 4096] + bias -> gate preactivations.
// 1024 waves (256 blocks x 4 waves); each wave: one M-tile x four N-tiles.
// K-loop FULLY UNROLLED (pure SSA -> no phi copies / hazard NOPs) with an
// enforced prefetch distance of one k-tile: tile kt+1's 10-load clause is
// issued (fenced) before tile kt's 4 WMMAs, so each WMMA group consumes
// loads that have had a full tile of work (~10 load issues + 4 WMMAs) to
// complete instead of draining loadcnt to 0 just-in-time.
// ---------------------------------------------------------------------------
__global__ void __launch_bounds__(128)
k_gemm(const bf16_t* __restrict__ Apack, const bf16_t* __restrict__ Bpack,
       const float* __restrict__ bias, float* __restrict__ gates) {
    int wave = blockIdx.x * 4 + (threadIdx.x >> 5);  // 0..1023
    int lane = threadIdx.x & 31;
    int mt   = wave >> 6;                            // 0..15
    int ntb  = (wave & 63) << 2;                     // 0..252 step 4

    const bf16_t* Ap = Apack + ((mt * KT) << 9) + (lane << 4);
    const bf16_t* Bp = Bpack + (ntb << 9) + (lane << 4);

    v8f acc[4] = {};

    // prologue: tile 0 fragments
    v16bf aC = *(const v16bf*)(Ap);
    v16bf bC[4];
#pragma unroll
    for (int j = 0; j < 4; ++j)
        bC[j] = *(const v16bf*)(Bp + (j << 9));

#pragma unroll
    for (int kt = 0; kt < KT; ++kt) {
        v16bf aN;
        v16bf bN[4];
        if (kt + 1 < KT) {   // compile-time folded per unrolled step
            aN = *(const v16bf*)(Ap + ((kt + 1) << 9));
#pragma unroll
            for (int j = 0; j < 4; ++j)
                bN[j] = *(const v16bf*)(Bp + (((kt + 1) * NTL + j) << 9));
        }
        SCHED_FENCE();  // keep next-tile loads ABOVE this tile's WMMAs
#pragma unroll
        for (int j = 0; j < 4; ++j)
            acc[j] = __builtin_amdgcn_wmma_f32_16x16x32_bf16(
                false, aC, false, bC[j], (short)0, acc[j], false, false);
        SCHED_FENCE();
        // SSA renames (fold away at unroll time; no machine copies)
        aC = aN;
#pragma unroll
        for (int j = 0; j < 4; ++j) bC[j] = bN[j];
    }

    // C/D layout: lane 0-15: N=lane, M=r; lane 16-31: N=lane-16, M=8+r
    int nloc = lane & 15;
    int mh   = lane >> 4;
#pragma unroll
    for (int j = 0; j < 4; ++j) {
        int ncol = ((ntb + j) << 4) + nloc;
        float bv = bias[ncol];
#pragma unroll
        for (int r = 0; r < 8; ++r) {
            int mrow = (mt << 4) + (mh << 3) + r;
            gates[mrow * NTOT + ncol] = acc[j][r] + bv;
        }
    }
}

// ---------------------------------------------------------------------------
// Elementwise gates + state update; writes h back into A-fragment region and
// (fused) packs x_{t+1} into the A-fragment x region for the next step.
// ---------------------------------------------------------------------------
__global__ void k_update(const float* __restrict__ gates, float* __restrict__ c,
                         float* __restrict__ h, bf16_t* __restrict__ Apack,
                         const float* __restrict__ inputs, int t_next) {
    int idx = blockIdx.x * 256 + threadIdx.x;  // BATCH*HSZ = 262144 threads
    int b = idx >> 10;
    int n = idx & 1023;
    const float* gr = gates + b * NTOT + n;
    float pi = gr[0];
    float pf = gr[HSZ];
    float pc = gr[2 * HSZ];
    float po = gr[3 * HSZ];
    float ig = 1.0f / (1.0f + expf(-pi));
    float fg = 1.0f / (1.0f + expf(-pf));
    float gg = tanhf(pc);
    float og = 1.0f / (1.0f + expf(-po));
    float cn = fg * c[idx] + ig * gg;
    c[idx] = cn;
    float hn = og * tanhf(cn);
    h[idx] = hn;
    // h -> A fragments at K = 512 + n for next step's GEMM
    Apack[a_frag_index(b, ISZ + n)] = (bf16_t)hn;
    // fused: pack x_{t+1} (first half of the threads cover BATCH*ISZ elems)
    if (t_next < SEQ && idx < BATCH * ISZ) {
        int xb = idx >> 9;
        int xk = idx & 511;
        Apack[a_frag_index(xb, xk)] =
            (bf16_t)inputs[(xb * SEQ + t_next) * ISZ + xk];
    }
}

// ---------------------------------------------------------------------------
// FC (1024 -> 32) + log_softmax. One block per batch row.
// ---------------------------------------------------------------------------
__global__ void k_fc(const float* __restrict__ h, const float* __restrict__ wfc,
                     float* __restrict__ out) {
    __shared__ float part[OSZ][8];
    __shared__ float red[OSZ];
    __shared__ float stats[2];
    int b = blockIdx.x;
    int o = threadIdx.x >> 3;
    int j = threadIdx.x & 7;
    const float* hr = h + b * HSZ;
    const float* wr = wfc + o * HSZ;
    float s = 0.0f;
    for (int k = j; k < HSZ; k += 8) s += hr[k] * wr[k];
    part[o][j] = s;
    __syncthreads();
    if (j == 0) {
        float t = 0.0f;
        for (int q = 0; q < 8; ++q) t += part[o][q];
        red[o] = t;
    }
    __syncthreads();
    if (threadIdx.x == 0) {
        float m = red[0];
        for (int q = 1; q < OSZ; ++q) m = fmaxf(m, red[q]);
        float sum = 0.0f;
        for (int q = 0; q < OSZ; ++q) sum += expf(red[q] - m);
        stats[0] = m;
        stats[1] = logf(sum);
    }
    __syncthreads();
    if (threadIdx.x < OSZ) {
        out[b * OSZ + threadIdx.x] = red[threadIdx.x] - stats[0] - stats[1];
    }
}

// ---------------------------------------------------------------------------
extern "C" void kernel_launch(void* const* d_in, const int* in_sizes, int n_in,
                              void* d_out, int out_size, void* d_ws, size_t ws_size,
                              hipStream_t stream) {
    const float* inputs = (const float*)d_in[0];
    const float* w_ii = (const float*)d_in[1];
    const float* w_hi = (const float*)d_in[2];
    const float* b_ii = (const float*)d_in[3];
    const float* b_hi = (const float*)d_in[4];
    const float* w_if = (const float*)d_in[5];
    const float* w_hf = (const float*)d_in[6];
    const float* b_if = (const float*)d_in[7];
    const float* b_hf = (const float*)d_in[8];
    const float* w_io = (const float*)d_in[9];
    const float* w_ho = (const float*)d_in[10];
    const float* b_io = (const float*)d_in[11];
    const float* b_ho = (const float*)d_in[12];
    const float* w_ic = (const float*)d_in[13];
    const float* w_hc = (const float*)d_in[14];
    const float* b_ic = (const float*)d_in[15];
    const float* b_hc = (const float*)d_in[16];
    const float* w_fc = (const float*)d_in[17];

    char* ws = (char*)d_ws;
    bf16_t* Bpack = (bf16_t*)(ws);                        // 12,582,912 B
    float*  bias  = (float*) (ws + 12582912);             //     16,384 B
    bf16_t* Apack = (bf16_t*)(ws + 12599296);             //    786,432 B
    float*  gates = (float*) (ws + 13385728);             //  4,194,304 B
    float*  cbuf  = (float*) (ws + 17580032);             //  1,048,576 B
    float*  hbuf  = (float*) (ws + 18628608);             //  1,048,576 B

    // one-time (per launch) packing + state init
    k_pack_weights<<<24576, 256, 0, stream>>>(w_ii, w_hi, w_if, w_hf,
                                              w_ic, w_hc, w_io, w_ho, Bpack);
    k_pack_bias<<<16, 256, 0, stream>>>(b_ii, b_hi, b_if, b_hf,
                                        b_ic, b_hc, b_io, b_ho, bias);
    k_init<<<1536, 256, 0, stream>>>(Apack, cbuf);
    k_pack_x<<<512, 256, 0, stream>>>(inputs, Apack, 0);   // prime x_0

    for (int t = 0; t < SEQ; ++t) {
        k_gemm<<<256, 128, 0, stream>>>(Apack, Bpack, bias, gates);
        k_update<<<1024, 256, 0, stream>>>(gates, cbuf, hbuf, Apack, inputs, t + 1);
    }

    k_fc<<<BATCH, 256, 0, stream>>>(hbuf, w_fc, (float*)d_out);
}
